// MemoryManagedDualAttentionModule_89404039233601
// MI455X (gfx1250) — compile-verified, matrix operand order checked by reading, not run on hardware
//
#include <hip/hip_runtime.h>
#include <math.h>

#define IMG   128
#define NPIX  (IMG*IMG)        // 16384
#define NH    8
#define DH    32
#define PD    768
#define KSPLIT 8
#define KCHUNK 2048
#define QBLK   256
#define USE_ASYNC_LDS 1

typedef _Float16 v16h __attribute__((ext_vector_type(16)));
typedef _Float16 h8v  __attribute__((ext_vector_type(8)));
typedef _Float16 h4v  __attribute__((ext_vector_type(4)));
typedef float    v8f  __attribute__((ext_vector_type(8)));

__device__ inline h4v cvt4(float4 v) {
  h4v r; r[0] = (_Float16)v.x; r[1] = (_Float16)v.y;
  r[2] = (_Float16)v.z; r[3] = (_Float16)v.w; return r;
}

// ---------------------------------------------------------------------------
// 3x3 conv, 3->3 channels, pad 1, NCHW/OIHW; optional exact GELU on output
// ---------------------------------------------------------------------------
__global__ void conv3x3_k(const float* __restrict__ in, const float* __restrict__ w,
                          float* __restrict__ out, int apply_gelu) {
  int idx = blockIdx.x * blockDim.x + threadIdx.x;
  if (idx >= NPIX) return;
  int y = idx >> 7, x = idx & 127;
  float acc0 = 0.f, acc1 = 0.f, acc2 = 0.f;
  #pragma unroll
  for (int ic = 0; ic < 3; ++ic) {
    #pragma unroll
    for (int dy = -1; dy <= 1; ++dy) {
      int yy = y + dy; if (yy < 0 || yy >= IMG) continue;
      #pragma unroll
      for (int dx = -1; dx <= 1; ++dx) {
        int xx = x + dx; if (xx < 0 || xx >= IMG) continue;
        float v = in[ic * NPIX + yy * IMG + xx];
        int wb = (ic * 3 + (dy + 1)) * 3 + (dx + 1);  // [oc][ic][ky][kx]
        acc0 += v * w[0 * 27 + wb];
        acc1 += v * w[1 * 27 + wb];
        acc2 += v * w[2 * 27 + wb];
      }
    }
  }
  float a[3] = {acc0, acc1, acc2};
  #pragma unroll
  for (int oc = 0; oc < 3; ++oc) {
    float v = a[oc];
    if (apply_gelu) v = 0.5f * v * (1.0f + erff(v * 0.70710678118654752f));
    out[oc * NPIX + idx] = v;
  }
}

// ---------------------------------------------------------------------------
// Per-pixel LayerNorm over 3 channels -> padded float4 token buffer sn
// ---------------------------------------------------------------------------
__global__ void seq_ln_k(const float* __restrict__ pe, const float* __restrict__ g,
                         const float* __restrict__ b, float* __restrict__ sn) {
  int q = blockIdx.x * blockDim.x + threadIdx.x;
  if (q >= NPIX) return;
  float c0 = pe[q], c1 = pe[NPIX + q], c2 = pe[2 * NPIX + q];
  float m = (c0 + c1 + c2) * (1.f / 3.f);
  float d0 = c0 - m, d1 = c1 - m, d2 = c2 - m;
  float v = (d0 * d0 + d1 * d1 + d2 * d2) * (1.f / 3.f);
  float r = rsqrtf(v + 1e-5f);
  float4 o;
  o.x = d0 * r * g[0] + b[0];
  o.y = d1 * r * g[1] + b[1];
  o.z = d2 * r * g[2] + b[2];
  o.w = 0.f;
  ((float4*)sn)[q] = o;
}

// ---------------------------------------------------------------------------
// Per-head 3x3 matrices: M_h = Wq_h^T Wk_h / sqrt(32), C_h = proj_s_h * Wv_h
// hm[0..71] = M, hm[72..143] = C
// ---------------------------------------------------------------------------
__global__ void headmats_k(const float* __restrict__ qkv_s_w,
                           const float* __restrict__ proj_s_w,
                           float* __restrict__ hm) {
  int t = threadIdx.x;
  if (t < 72) {
    int h = t / 9, r = t % 9, i = r / 3, j = r % 3;
    float s = 0.f;
    for (int d = 0; d < DH; ++d)
      s += qkv_s_w[(h * DH + d) * 3 + i] * qkv_s_w[(256 + h * DH + d) * 3 + j];
    hm[t] = s * 0.17677669529663689f;  // 1/sqrt(32)
  } else if (t < 144) {
    int u = t - 72;
    int h = u / 9, r = u % 9, c = r / 3, j = r % 3;
    float s = 0.f;
    for (int d = 0; d < DH; ++d)
      s += proj_s_w[c * 256 + h * DH + d] * qkv_s_w[(512 + h * DH + d) * 3 + j];
    hm[72 + u] = s;  // hm[72 + h*9 + c*3 + j]
  }
}

// ---------------------------------------------------------------------------
// Rank-3 flash attention, key-split partials (no max shift: scores are O(1),
// softmax shift-invariant -> partials combine by addition).
// Key staging uses gfx1250 GLOBAL_LOAD_ASYNC_TO_LDS_B128 (ASYNCcnt).
// part[((sp*NPIX + q)*8 + h)] = float4(l, t0, t1, t2)
// ---------------------------------------------------------------------------
__global__ __launch_bounds__(QBLK)
void flash_seq_k(const float* __restrict__ sn, const float* __restrict__ hm,
                 float* __restrict__ part) {
  __shared__ float4 kb[KCHUNK];
  int q  = blockIdx.x * QBLK + threadIdx.x;
  int sp = blockIdx.y;
  int k0 = sp * KCHUNK;

#if USE_ASYNC_LDS
  {
    unsigned long long gbase = (unsigned long long)(const void*)(((const float4*)sn) + k0);
    #pragma unroll
    for (int it = 0; it < KCHUNK / QBLK; ++it) {
      int i = it * QBLK + threadIdx.x;
      unsigned int  laddr = (unsigned int)(unsigned long long)(&kb[i]);
      unsigned long long gaddr = gbase + (unsigned long long)i * 16ull;
      asm volatile("global_load_async_to_lds_b128 %0, %1, off"
                   :: "v"(laddr), "v"(gaddr) : "memory");
    }
    asm volatile("s_wait_asynccnt 0" ::: "memory");
  }
#else
  for (int i = threadIdx.x; i < KCHUNK; i += QBLK)
    kb[i] = ((const float4*)sn)[k0 + i];
#endif
  __syncthreads();

  float4 sq = ((const float4*)sn)[q];
  float a0[NH], a1[NH], a2[NH], l[NH], t0[NH], t1[NH], t2[NH];
  #pragma unroll
  for (int h = 0; h < NH; ++h) {
    const float* M = hm + h * 9;           // a = M^T * sn_q
    a0[h] = sq.x * M[0] + sq.y * M[3] + sq.z * M[6];
    a1[h] = sq.x * M[1] + sq.y * M[4] + sq.z * M[7];
    a2[h] = sq.x * M[2] + sq.y * M[5] + sq.z * M[8];
    l[h] = 0.f; t0[h] = 0.f; t1[h] = 0.f; t2[h] = 0.f;
  }
  for (int j = 0; j < KCHUNK; ++j) {
    float4 kv = kb[j];                     // uniform address -> LDS broadcast
    #pragma unroll
    for (int h = 0; h < NH; ++h) {
      float s = a0[h] * kv.x + a1[h] * kv.y + a2[h] * kv.z;
      float e = __expf(s);
      l[h]  += e;
      t0[h] += e * kv.x;
      t1[h] += e * kv.y;
      t2[h] += e * kv.z;
    }
  }
  float4* P = (float4*)part + ((size_t)sp * NPIX + q) * NH;
  #pragma unroll
  for (int h = 0; h < NH; ++h) {
    float4 o; o.x = l[h]; o.y = t0[h]; o.z = t1[h]; o.w = t2[h];
    P[h] = o;
  }
}

// Combine splits, apply folded C_h = proj_s*Wv, add proj_s bias -> out_seq [3,NPIX]
__global__ void seq_out_k(const float* __restrict__ part, const float* __restrict__ hm,
                          const float* __restrict__ proj_s_b, float* __restrict__ outseq) {
  int q = blockIdx.x * blockDim.x + threadIdx.x;
  if (q >= NPIX) return;
  float r0 = proj_s_b[0], r1 = proj_s_b[1], r2 = proj_s_b[2];
  #pragma unroll
  for (int h = 0; h < NH; ++h) {
    float l = 0.f, t0 = 0.f, t1 = 0.f, t2 = 0.f;
    for (int sp = 0; sp < KSPLIT; ++sp) {
      float4 v = ((const float4*)part)[((size_t)sp * NPIX + q) * NH + h];
      l += v.x; t0 += v.y; t1 += v.z; t2 += v.w;
    }
    float inv = 1.f / l;
    t0 *= inv; t1 *= inv; t2 *= inv;
    const float* C = hm + 72 + h * 9;
    r0 += C[0] * t0 + C[1] * t1 + C[2] * t2;
    r1 += C[3] * t0 + C[4] * t1 + C[5] * t2;
    r2 += C[6] * t0 + C[7] * t1 + C[8] * t2;
  }
  outseq[q] = r0; outseq[NPIX + q] = r1; outseq[2 * NPIX + q] = r2;
}

// ---------------------------------------------------------------------------
// Gather pe into patch-major matrix A[64][768], col = i*256 + p*16 + q
// ---------------------------------------------------------------------------
__global__ void patch_gather_k(const float* __restrict__ pe, float* __restrict__ A) {
  int idx = blockIdx.x * blockDim.x + threadIdx.x;
  if (idx >= 64 * PD) return;
  int patch = idx / PD, col = idx % PD;
  int i = col >> 8, rem = col & 255, p = rem >> 4, qq = rem & 15;
  int ph = patch >> 3, pw = patch & 7;
  A[idx] = pe[i * NPIX + (ph * 16 + p) * IMG + pw * 16 + qq];
}

// ---------------------------------------------------------------------------
// LayerNorm over 768 features, one block per row
// ---------------------------------------------------------------------------
__global__ void patch_ln_k(const float* __restrict__ src, const float* __restrict__ g,
                           const float* __restrict__ b, float* __restrict__ dst) {
  __shared__ float red[256];
  int row = blockIdx.x;
  const float* x = src + row * PD;
  float s = 0.f;
  for (int i = threadIdx.x; i < PD; i += 256) s += x[i];
  red[threadIdx.x] = s; __syncthreads();
  for (int o = 128; o > 0; o >>= 1) { if (threadIdx.x < o) red[threadIdx.x] += red[threadIdx.x + o]; __syncthreads(); }
  float mean = red[0] * (1.f / PD);
  __syncthreads();
  float v = 0.f;
  for (int i = threadIdx.x; i < PD; i += 256) { float d = x[i] - mean; v += d * d; }
  red[threadIdx.x] = v; __syncthreads();
  for (int o = 128; o > 0; o >>= 1) { if (threadIdx.x < o) red[threadIdx.x] += red[threadIdx.x + o]; __syncthreads(); }
  float r = rsqrtf(red[0] * (1.f / PD) + 1e-5f);
  for (int i = threadIdx.x; i < PD; i += 256)
    dst[row * PD + i] = (x[i] - mean) * r * g[i] + b[i];
}

// ---------------------------------------------------------------------------
// WMMA GEMM: C[64,768] = A[64,K] * B (+bias[n]).  K is compile-time.
// BT=0: B row-major [K,768]; BT=1: B is [768,K] (transposed access).
// 128 threads = 4 waves; block tile 16(M) x 64(N); f16 operands, f32 accum.
// LDS tiles stored k-contiguous per row (A[m][k], Bt[n][k]) with rows padded
// to 40 halves (80B): fragments load as 2x ds_load_b128 each, bank-conflict
// free (lane addr = L*20+g*4 mod 64, all distinct).
// ---------------------------------------------------------------------------
template <int BT, int BIAS, int K>
__global__ __launch_bounds__(128)
void wmma_gemm_k(const float* __restrict__ A, const float* __restrict__ B,
                 const float* __restrict__ bias, float* __restrict__ C) {
  constexpr int N = 768;
  __shared__ _Float16 As[16][40];   // [m][k], padded row
  __shared__ _Float16 Bt[64][40];   // [n][k], padded row
  const int m0 = blockIdx.x * 16, n0 = blockIdx.y * 64;
  const int tid = threadIdx.x;
  const int lane = tid & 31, wave = tid >> 5;
  const int ln = lane & 15, g = lane >> 4;

  // fixed per-thread staging coordinates (affine, compile-time strides)
  const int ar = tid >> 3;          // A row 0..15
  const int ac = (tid & 7) * 4;     // A col quad base
  const int bc = tid >> 1;          // BT=1: B col 0..63
  const int bk = (tid & 1) * 16;    // BT=1: k base {0,16}
  const int br = tid >> 2;          // BT=0: k row 0..31
  const int bn = (tid & 3) * 16;    // BT=0: n base {0,16,32,48}

  v8f acc = {};
  for (int kk = 0; kk < K; kk += 32) {
    // ---- load globals into registers (clause-able, one wait) ----
    float4 av = *(const float4*)&A[(size_t)(m0 + ar) * K + kk + ac];
    float4 b0, b1, b2, b3;
    if (BT) {
      const float* Bp = &B[(size_t)(n0 + bc) * K + kk + bk];
      b0 = *(const float4*)(Bp + 0);
      b1 = *(const float4*)(Bp + 4);
      b2 = *(const float4*)(Bp + 8);
      b3 = *(const float4*)(Bp + 12);
    } else {
      const float* Bp = &B[(size_t)(kk + br) * N + n0 + bn];
      b0 = *(const float4*)(Bp + 0);
      b1 = *(const float4*)(Bp + 4);
      b2 = *(const float4*)(Bp + 8);
      b3 = *(const float4*)(Bp + 12);
    }
    __syncthreads();                 // previous tile fully consumed
    // ---- convert + stage to LDS ----
    *(h4v*)&As[ar][ac] = cvt4(av);
    if (BT) {
      *(h4v*)&Bt[bc][bk + 0]  = cvt4(b0);
      *(h4v*)&Bt[bc][bk + 4]  = cvt4(b1);
      *(h4v*)&Bt[bc][bk + 8]  = cvt4(b2);
      *(h4v*)&Bt[bc][bk + 12] = cvt4(b3);
    } else {
      h4v c0 = cvt4(b0), c1 = cvt4(b1), c2 = cvt4(b2), c3 = cvt4(b3);
      #pragma unroll
      for (int e = 0; e < 4; ++e) {
        Bt[bn + 0  + e][br] = c0[e];
        Bt[bn + 4  + e][br] = c1[e];
        Bt[bn + 8  + e][br] = c2[e];
        Bt[bn + 12 + e][br] = c3[e];
      }
    }
    __syncthreads();                 // stage visible
    // ---- fragments: 2x b128 per operand (ISA 7.12.2 16-bit layout) ----
    h8v alo = *(const h8v*)&As[ln][g * 8];
    h8v ahi = *(const h8v*)&As[ln][16 + g * 8];
    h8v blo = *(const h8v*)&Bt[wave * 16 + ln][g * 8];
    h8v bhi = *(const h8v*)&Bt[wave * 16 + ln][16 + g * 8];
    v16h af, bf;
    #pragma unroll
    for (int i = 0; i < 8; ++i) {
      af[i] = alo[i]; af[8 + i] = ahi[i];
      bf[i] = blo[i]; bf[8 + i] = bhi[i];
    }
    acc = __builtin_amdgcn_wmma_f32_16x16x32_f16(false, af, false, bf,
                                                 (short)0, acc, false, false);
  }
  #pragma unroll
  for (int r = 0; r < 8; ++r) {
    int m = m0 + g * 8 + r;
    int n = n0 + wave * 16 + ln;
    float v = acc[r];
    if (BIAS) v += bias[n];
    C[(size_t)m * N + n] = v;
  }
}

// ---------------------------------------------------------------------------
// Patch attention: 64 tokens, one block (64 threads) per head, online softmax
// ---------------------------------------------------------------------------
__global__ void patch_attn_k(const float* __restrict__ qkv, float* __restrict__ o) {
  int h = blockIdx.x;
  __shared__ float kb[64][DH], vb[64][DH];
  int t = threadIdx.x;  // one query row per thread
  #pragma unroll
  for (int d = 0; d < DH; ++d) {
    kb[t][d] = qkv[t * PD + 256 + h * DH + d];
    vb[t][d] = qkv[t * PD + 512 + h * DH + d];
  }
  float qr[DH];
  #pragma unroll
  for (int d = 0; d < DH; ++d) qr[d] = qkv[t * PD + h * DH + d];
  __syncthreads();
  float m = -1e30f, l = 0.f, acc[DH];
  #pragma unroll
  for (int d = 0; d < DH; ++d) acc[d] = 0.f;
  for (int j = 0; j < 64; ++j) {
    float s = 0.f;
    #pragma unroll
    for (int d = 0; d < DH; ++d) s += qr[d] * kb[j][d];
    s *= 0.17677669529663689f;
    float nm = fmaxf(m, s);
    float cor = __expf(m - nm), e = __expf(s - nm);
    l = l * cor + e;
    #pragma unroll
    for (int d = 0; d < DH; ++d) acc[d] = acc[d] * cor + e * vb[j][d];
    m = nm;
  }
  float inv = 1.f / l;
  #pragma unroll
  for (int d = 0; d < DH; ++d) o[t * 256 + h * DH + d] = acc[d] * inv;
}

// ---------------------------------------------------------------------------
// combined = fold(patC) + rec_b + out_seq  (image layout [3,128,128])
// ---------------------------------------------------------------------------
__global__ void combine_k(const float* __restrict__ patC, const float* __restrict__ rec_b,
                          const float* __restrict__ outseq, float* __restrict__ comb) {
  int idx = blockIdx.x * blockDim.x + threadIdx.x;
  if (idx >= 3 * NPIX) return;
  int c = idx / NPIX, pix = idx % NPIX;
  int y = pix >> 7, x = pix & 127;
  int patch = (y >> 4) * 8 + (x >> 4);
  int col = c * 256 + (y & 15) * 16 + (x & 15);
  comb[idx] = patC[(size_t)patch * PD + col] + rec_b[c] + outseq[idx];
}

// ---------------------------------------------------------------------------
extern "C" void kernel_launch(void* const* d_in, const int* in_sizes, int n_in,
                              void* d_out, int out_size, void* d_ws, size_t ws_size,
                              hipStream_t stream) {
  (void)in_sizes; (void)n_in; (void)out_size; (void)ws_size;
  const float* x        = (const float*)d_in[0];
  const float* pe_w1    = (const float*)d_in[1];
  const float* pe_w2    = (const float*)d_in[2];
  const float* patch_w  = (const float*)d_in[3];
  const float* patch_b  = (const float*)d_in[4];
  const float* qkv_p_w  = (const float*)d_in[5];
  const float* qkv_s_w  = (const float*)d_in[6];
  const float* proj_p_w = (const float*)d_in[7];
  const float* proj_p_b = (const float*)d_in[8];
  const float* proj_s_w = (const float*)d_in[9];
  const float* proj_s_b = (const float*)d_in[10];
  const float* rec_w    = (const float*)d_in[11];
  const float* rec_b    = (const float*)d_in[12];
  const float* pd_w1    = (const float*)d_in[13];
  const float* pd_w2    = (const float*)d_in[14];
  const float* ln1_g    = (const float*)d_in[15];
  const float* ln1_b    = (const float*)d_in[16];
  const float* ln2_g    = (const float*)d_in[17];
  const float* ln2_b    = (const float*)d_in[18];
  float* out = (float*)d_out;

  float* w = (float*)d_ws;
  float* tmp     = w; w += 3 * NPIX;        // conv scratch
  float* pe      = w; w += 3 * NPIX;        // pos-encoded image
  float* sn      = w; w += 4 * NPIX;        // LN'd tokens, float4-padded
  float* Apat    = w; w += 64 * PD;         // gathered patch inputs
  float* patches = w; w += 64 * PD;
  float* pn      = w; w += 64 * PD;
  float* qkv     = w; w += 64 * PD;
  float* attno   = w; w += 64 * 256;
  float* oprj    = w; w += 64 * PD;
  float* patC    = w; w += 64 * PD;
  float* outseq  = w; w += 3 * NPIX;
  float* comb    = w; w += 3 * NPIX;
  float* hm      = w; w += 256;             // 72 M + 72 C
  float* part    = w; w += (size_t)KSPLIT * NPIX * NH * 4;

  // pos_encoding
  conv3x3_k<<<NPIX / 256, 256, 0, stream>>>(x,   pe_w1, tmp, 1);
  conv3x3_k<<<NPIX / 256, 256, 0, stream>>>(tmp, pe_w2, pe,  0);

  // ---- sequence branch (rank-3 factorized flash attention) ----
  seq_ln_k<<<NPIX / 256, 256, 0, stream>>>(pe, ln2_g, ln2_b, sn);
  headmats_k<<<1, 160, 0, stream>>>(qkv_s_w, proj_s_w, hm);
  flash_seq_k<<<dim3(NPIX / QBLK, KSPLIT), QBLK, 0, stream>>>(sn, hm, part);
  seq_out_k<<<NPIX / 256, 256, 0, stream>>>(part, hm, proj_s_b, outseq);

  // ---- patch branch (WMMA GEMMs) ----
  patch_gather_k<<<(64 * PD) / 256, 256, 0, stream>>>(pe, Apat);
  wmma_gemm_k<1, 1, 768><<<dim3(4, 12), 128, 0, stream>>>(Apat, patch_w, patch_b, patches);
  patch_ln_k<<<64, 256, 0, stream>>>(patches, ln1_g, ln1_b, pn);
  wmma_gemm_k<1, 0, 768><<<dim3(4, 12), 128, 0, stream>>>(pn, qkv_p_w, nullptr, qkv);
  patch_attn_k<<<NH, 64, 0, stream>>>(qkv, attno);
  wmma_gemm_k<1, 1, 256><<<dim3(4, 12), 128, 0, stream>>>(attno, proj_p_w, proj_p_b, oprj);
  wmma_gemm_k<0, 0, 768><<<dim3(4, 12), 128, 0, stream>>>(oprj, rec_w, nullptr, patC);

  // combine + pos_decoding
  combine_k<<<(3 * NPIX) / 256, 256, 0, stream>>>(patC, rec_b, outseq, comb);
  conv3x3_k<<<NPIX / 256, 256, 0, stream>>>(comb, pd_w1, tmp, 1);
  conv3x3_k<<<NPIX / 256, 256, 0, stream>>>(tmp, pd_w2, out, 0);
}
